// MolEncoder_84078279786623
// MI455X (gfx1250) — matrix operand order repaired and checked by reading, not compiled, not run on hardware
//
#include <hip/hip_runtime.h>
#include <hip/hip_bf16.h>

typedef __attribute__((ext_vector_type(16))) __bf16 v16bf;
typedef __attribute__((ext_vector_type(8)))  float  v8f;

constexpr int NN    = 100000;
constexpr int EE    = 640000;
constexpr int BB    = 2000;
constexpr int NDIM  = 256;
constexpr int EDIM  = 128;
constexpr int LYRS  = 3;
constexpr int EMBD  = 256;

// ---------- helpers ----------
__device__ __forceinline__ __bf16 f2bf(float f) {
  union { float f; unsigned u; } v; v.f = f;
  unsigned r = v.u + 0x7FFFu + ((v.u >> 16) & 1u);
  union { unsigned short s; __bf16 b; } o; o.s = (unsigned short)(r >> 16);
  return o.b;
}

// A-fragment (16x32 bf16) from LDS row-major panel. Layout per ISA 7.12.2:
// lane r (0..15) row M=r holds K {0..7,16..23}; lanes 16..31 hold K {8..15,24..31}.
__device__ __forceinline__ v16bf lds_a_frag(const __bf16* base, int stride,
                                            int m0, int k0, int lane) {
  int r = lane & 15, h = lane >> 4;
  const __bf16* p = base + (m0 + r) * stride + k0 + h * 8;
  union { uint4 u[2]; v16bf v; } c;
  c.u[0] = *(const uint4*)(p);       // K  0..7  (of this half)
  c.u[1] = *(const uint4*)(p + 16);  // K 16..23
  return c.v;
}

// B-fragment (32x16 bf16) from pre-packed weights: one contiguous 32B per lane.
__device__ __forceinline__ v16bf ldg_b_frag(const __bf16* wpk, int ktiles,
                                            int kt, int nt, int lane) {
  const __bf16* p = wpk + ((size_t)(nt * ktiles + kt) << 9) + (lane << 4);
  union { uint4 u[2]; v16bf v; } c;
  c.u[0] = *(const uint4*)(p);
  c.u[1] = *(const uint4*)(p + 8);
  return c.v;
}

__device__ __forceinline__ v8f wmma_bf16(v16bf a, v16bf b, v8f c) {
  return __builtin_amdgcn_wmma_f32_16x16x32_bf16(false, a, false, b,
                                                 (short)0, c, false, false);
}

// ---------- weight packing: fp32 [K,Ncols] -> bf16 B-fragment tiles ----------
__global__ __launch_bounds__(256) void pack_w_kernel(
    const float* __restrict__ W, __bf16* __restrict__ out, int K, int Ncols) {
  int tid = blockIdx.x * blockDim.x + threadIdx.x;
  int lane = tid & 31;
  int tile = tid >> 5;
  int ktiles = K >> 5, ntiles = Ncols >> 4;
  if (tile >= ktiles * ntiles) return;
  int kt = tile % ktiles, nt = tile / ktiles;
  int col  = nt * 16 + (lane & 15);
  int krow = kt * 32 + (lane >> 4) * 8;
  __bf16* o = out + ((size_t)(nt * ktiles + kt) << 9) + (lane << 4);
#pragma unroll
  for (int j = 0; j < 16; ++j)
    o[j] = f2bf(W[(size_t)(krow + (j & 7) + ((j >> 3) << 4)) * Ncols + col]);
}

// ---------- embeddings ----------
__global__ __launch_bounds__(256) void embed_node_kernel(
    const float* __restrict__ hN, const float* __restrict__ Wn,
    float* __restrict__ hnf, __bf16* __restrict__ hnb) {
  __shared__ float sH[32][16];
  const int tid = threadIdx.x;
  const int n0 = blockIdx.x * 32;
  for (int idx = tid; idx < 512; idx += 256)
    sH[idx >> 4][idx & 15] = hN[(size_t)n0 * 16 + idx];
  __syncthreads();
  int row = tid >> 3, c0 = tid & 7;
#pragma unroll 4
  for (int j = 0; j < 32; ++j) {
    int col = c0 + 8 * j;
    float s = 0.f;
#pragma unroll
    for (int k = 0; k < 16; ++k) s += sH[row][k] * Wn[k * NDIM + col];
    size_t go = (size_t)(n0 + row) * NDIM + col;
    hnf[go] = s; hnb[go] = f2bf(s);
  }
}

__global__ __launch_bounds__(256) void embed_edge_kernel(
    const float* __restrict__ hE, const float* __restrict__ We,
    const float* __restrict__ pos, const int* __restrict__ src,
    const int* __restrict__ dstp, float* __restrict__ hef,
    __bf16* __restrict__ heb, float* __restrict__ distg) {
  __shared__ float sH[32][5];
  const int tid = threadIdx.x;
  const int e0 = blockIdx.x * 32;
  for (int idx = tid; idx < 160; idx += 256)
    sH[idx / 5][idx % 5] = hE[(size_t)e0 * 5 + idx];
  if (tid < 32) {
    int e = e0 + tid;
    int s = src[e], d = dstp[e];
    float dx = pos[d * 3 + 0] - pos[s * 3 + 0];
    float dy = pos[d * 3 + 1] - pos[s * 3 + 1];
    float dz = pos[d * 3 + 2] - pos[s * 3 + 2];
    distg[e] = sqrtf(dx * dx + dy * dy + dz * dz);
  }
  __syncthreads();
  int row = tid >> 3, c0 = tid & 7;
#pragma unroll 4
  for (int j = 0; j < 16; ++j) {
    int col = c0 + 8 * j;
    float s = 0.f;
#pragma unroll
    for (int k = 0; k < 5; ++k) s += sH[row][k] * We[k * EDIM + col];
    size_t go = (size_t)(e0 + row) * EDIM + col;
    hef[go] = s; heb[go] = f2bf(s);
  }
}

// ---------- fused edge block: gather -> GEMM1(+dist,relu) -> GEMM2(relu) -> scatter ----------
constexpr int SA_STR = 648;  // 640 + 8 pad (keeps 16B alignment, breaks bank clash)
constexpr int SH_STR = 136;  // 128 + 8 pad

__global__ __launch_bounds__(256) void edge_layer_kernel(
    const __bf16* __restrict__ hnb, __bf16* __restrict__ heb,
    float* __restrict__ hef, const float* __restrict__ distg,
    const int* __restrict__ src, const int* __restrict__ dstp,
    const __bf16* __restrict__ weu_pk, const float* __restrict__ weu_last,
    const float* __restrict__ beu, const __bf16* __restrict__ wmsg_pk,
    const float* __restrict__ bmsg, float* __restrict__ agg) {
  __shared__ __bf16 sA[32 * SA_STR];   // [he | hn_src | hn_dst], K=640
  __shared__ __bf16 sHe[32 * SH_STR];  // he' (post-relu, bf16)
  __shared__ float sDist[32];
  __shared__ int sSrc[32], sDst[32];

  const int tid = threadIdx.x;
  const int lane = tid & 31;
  const int wave = tid >> 5;
  const int e0 = blockIdx.x * 32;

  if (tid < 32) {
    int e = e0 + tid;
    sSrc[tid] = src[e];
    sDst[tid] = dstp[e];
    sDist[tid] = distg[e];
  }
  __syncthreads();

  // stage A-panel: 80 x 16B chunks per edge row
  for (int idx = tid; idx < 32 * 80; idx += 256) {
    int row = idx / 80, c = idx - row * 80;
    uint4 v; int off;
    if (c < 16) {
      v = *(const uint4*)(heb + (size_t)(e0 + row) * EDIM + c * 8);
      off = c * 8;
    } else if (c < 48) {
      v = *(const uint4*)(hnb + (size_t)sSrc[row] * NDIM + (c - 16) * 8);
      off = 128 + (c - 16) * 8;
    } else {
      v = *(const uint4*)(hnb + (size_t)sDst[row] * NDIM + (c - 48) * 8);
      off = 384 + (c - 48) * 8;
    }
    *(uint4*)(sA + row * SA_STR + off) = v;
  }
  __syncthreads();

  // GEMM1: he' = relu(e_in @ W_eu + dist*w640 + b)   (32x128, K=640)
  const int mt = wave >> 2;             // row tile 0..1
  const int nt0 = (wave & 3) * 2;       // col tiles nt0, nt0+1
  v8f a0 = {}, a1 = {};
  for (int kt = 0; kt < 20; ++kt) {
    v16bf a = lds_a_frag(sA, SA_STR, mt * 16, kt * 32, lane);
    v16bf b0 = ldg_b_frag(weu_pk, 20, kt, nt0, lane);
    v16bf b1 = ldg_b_frag(weu_pk, 20, kt, nt0 + 1, lane);
    a0 = wmma_bf16(a, b0, a0);
    a1 = wmma_bf16(a, b1, a1);
  }
  const int h = lane >> 4, cl = lane & 15;
#pragma unroll
  for (int t = 0; t < 2; ++t) {
    v8f acc = t ? a1 : a0;
    int col = (nt0 + t) * 16 + cl;
    float wl = weu_last[col];
    float bb = beu[col];
#pragma unroll
    for (int i = 0; i < 8; ++i) {
      int row = mt * 16 + i + 8 * h;
      float v = fmaxf(acc[i] + bb + sDist[row] * wl, 0.0f);
      sHe[row * SH_STR + col] = f2bf(v);
      size_t go = (size_t)(e0 + row) * EDIM + col;
      hef[go] = v;
      heb[go] = f2bf(v);
    }
  }
  __syncthreads();

  // GEMM2: msg = relu([hn_src | he'] @ W_msg + b)    (32x256, K=384)
  const int ntb = (wave & 3) * 4;
  v8f m0 = {}, m1 = {}, m2 = {}, m3 = {};
  for (int kt = 0; kt < 12; ++kt) {
    v16bf a = (kt < 8)
                  ? lds_a_frag(sA, SA_STR, mt * 16, 128 + kt * 32, lane)
                  : lds_a_frag(sHe, SH_STR, mt * 16, (kt - 8) * 32, lane);
    m0 = wmma_bf16(a, ldg_b_frag(wmsg_pk, 12, kt, ntb + 0, lane), m0);
    m1 = wmma_bf16(a, ldg_b_frag(wmsg_pk, 12, kt, ntb + 1, lane), m1);
    m2 = wmma_bf16(a, ldg_b_frag(wmsg_pk, 12, kt, ntb + 2, lane), m2);
    m3 = wmma_bf16(a, ldg_b_frag(wmsg_pk, 12, kt, ntb + 3, lane), m3);
  }
#pragma unroll
  for (int t = 0; t < 4; ++t) {
    v8f acc = (t == 0) ? m0 : (t == 1) ? m1 : (t == 2) ? m2 : m3;
    int col = (ntb + t) * 16 + cl;
    float bb = bmsg[col];
#pragma unroll
    for (int i = 0; i < 8; ++i) {
      int row = mt * 16 + i + 8 * h;
      float v = fmaxf(acc[i] + bb, 0.0f);
      atomicAdd(agg + (size_t)sDst[row] * NDIM + col, v);
    }
  }
}

// ---------- node update: hn += relu([hn|agg] @ W_nu + b) ----------
__global__ __launch_bounds__(256) void node_layer_kernel(
    float* __restrict__ hnf, __bf16* __restrict__ hnb,
    const float* __restrict__ agg, const __bf16* __restrict__ wnu_pk,
    const float* __restrict__ bnu) {
  __shared__ __bf16 sA[32 * 520];  // K=512 + 8 pad
  const int tid = threadIdx.x, lane = tid & 31, wave = tid >> 5;
  const int n0 = blockIdx.x * 32;

  for (int idx = tid; idx < 32 * 32; idx += 256) {
    int row = idx >> 5, c = idx & 31;
    *(uint4*)(sA + row * 520 + c * 8) =
        *(const uint4*)(hnb + (size_t)(n0 + row) * NDIM + c * 8);
  }
  for (int idx = tid; idx < 32 * 32; idx += 256) {
    int row = idx >> 5, c = idx & 31;
    const float* p = agg + (size_t)(n0 + row) * NDIM + c * 8;
    __align__(16) __bf16 tmp[8];
#pragma unroll
    for (int j = 0; j < 8; ++j) tmp[j] = f2bf(p[j]);
    *(uint4*)(sA + row * 520 + 256 + c * 8) = *(const uint4*)tmp;
  }
  __syncthreads();

  const int mt = wave >> 2, ntb = (wave & 3) * 4;
  v8f m0 = {}, m1 = {}, m2 = {}, m3 = {};
  for (int kt = 0; kt < 16; ++kt) {
    v16bf a = lds_a_frag(sA, 520, mt * 16, kt * 32, lane);
    m0 = wmma_bf16(a, ldg_b_frag(wnu_pk, 16, kt, ntb + 0, lane), m0);
    m1 = wmma_bf16(a, ldg_b_frag(wnu_pk, 16, kt, ntb + 1, lane), m1);
    m2 = wmma_bf16(a, ldg_b_frag(wnu_pk, 16, kt, ntb + 2, lane), m2);
    m3 = wmma_bf16(a, ldg_b_frag(wnu_pk, 16, kt, ntb + 3, lane), m3);
  }
  const int h = lane >> 4, cl = lane & 15;
#pragma unroll
  for (int t = 0; t < 4; ++t) {
    v8f acc = (t == 0) ? m0 : (t == 1) ? m1 : (t == 2) ? m2 : m3;
    int col = (ntb + t) * 16 + cl;
    float bb = bnu[col];
#pragma unroll
    for (int i = 0; i < 8; ++i) {
      int row = mt * 16 + i + 8 * h;
      size_t go = (size_t)(n0 + row) * NDIM + col;
      float v = hnf[go] + fmaxf(acc[i] + bb, 0.0f);
      hnf[go] = v;
      hnb[go] = f2bf(v);
    }
  }
}

// ---------- pooling ----------
__global__ __launch_bounds__(256) void pool_node_kernel(
    const float* __restrict__ hn, const int* __restrict__ bn,
    float* __restrict__ pool, float* __restrict__ cnt) {
  int tid = blockIdx.x * blockDim.x + threadIdx.x;
  int n = tid >> 6;
  if (n >= NN) return;
  int c = (tid & 63) * 4;
  int b = bn[n];
  const float* p = hn + (size_t)n * NDIM + c;
  float* q = pool + (size_t)b * 384 + c;
  atomicAdd(q + 0, p[0]); atomicAdd(q + 1, p[1]);
  atomicAdd(q + 2, p[2]); atomicAdd(q + 3, p[3]);
  if (c == 0) atomicAdd(cnt + b, 1.0f);
}

__global__ __launch_bounds__(256) void pool_edge_kernel(
    const float* __restrict__ he, const int* __restrict__ be,
    float* __restrict__ pool, float* __restrict__ cnt) {
  int tid = blockIdx.x * blockDim.x + threadIdx.x;
  int e = tid >> 5;
  if (e >= EE) return;
  int c = (tid & 31) * 4;
  int b = be[e];
  const float* p = he + (size_t)e * EDIM + c;
  float* q = pool + (size_t)b * 384 + 256 + c;
  atomicAdd(q + 0, p[0]); atomicAdd(q + 1, p[1]);
  atomicAdd(q + 2, p[2]); atomicAdd(q + 3, p[3]);
  if (c == 0) atomicAdd(cnt + b, 1.0f);
}

// ---------- fused final MLP: relu(h_sub@Wf1+b1)@Wf2+b2 ----------
__global__ __launch_bounds__(256) void final_mlp_kernel(
    const float* __restrict__ pool, const float* __restrict__ cntn,
    const float* __restrict__ cnte, const __bf16* __restrict__ wf1_pk,
    const float* __restrict__ bf1p, const __bf16* __restrict__ wf2_pk,
    const float* __restrict__ bf2p, float* __restrict__ out) {
  __shared__ __bf16 sA[32 * 392];    // K=384 + 8 pad
  __shared__ __bf16 sHid[32 * 520];  // K=512 + 8 pad
  __shared__ float sRn[32], sRe[32];
  const int tid = threadIdx.x, lane = tid & 31, wave = tid >> 5;
  const int b0 = blockIdx.x * 32;
  if (tid < 32) {
    int b = b0 + tid; if (b >= BB) b = BB - 1;
    sRn[tid] = 1.0f / fmaxf(cntn[b], 1.0f);
    sRe[tid] = 1.0f / fmaxf(cnte[b], 1.0f);
  }
  __syncthreads();
  for (int idx = tid; idx < 32 * 384; idx += 256) {
    int row = idx / 384, c = idx - row * 384;
    int b = b0 + row; if (b >= BB) b = BB - 1;
    float v = pool[(size_t)b * 384 + c] * (c < 256 ? sRn[row] : sRe[row]);
    sA[row * 392 + c] = f2bf(v);
  }
  __syncthreads();

  const int mt = wave >> 2, h = lane >> 4, cl = lane & 15;
  {  // GEMM1: 32x512, K=384
    const int ntb = (wave & 3) * 8;
    v8f acc[8] = {};
    for (int kt = 0; kt < 12; ++kt) {
      v16bf a = lds_a_frag(sA, 392, mt * 16, kt * 32, lane);
#pragma unroll
      for (int t = 0; t < 8; ++t)
        acc[t] = wmma_bf16(a, ldg_b_frag(wf1_pk, 12, kt, ntb + t, lane), acc[t]);
    }
#pragma unroll
    for (int t = 0; t < 8; ++t) {
      int col = (ntb + t) * 16 + cl;
      float bb = bf1p[col];
#pragma unroll
      for (int i = 0; i < 8; ++i) {
        int row = mt * 16 + i + 8 * h;
        sHid[row * 520 + col] = f2bf(fmaxf(acc[t][i] + bb, 0.0f));
      }
    }
  }
  __syncthreads();
  {  // GEMM2: 32x256, K=512
    const int ntb = (wave & 3) * 4;
    v8f m0 = {}, m1 = {}, m2 = {}, m3 = {};
    for (int kt = 0; kt < 16; ++kt) {
      v16bf a = lds_a_frag(sHid, 520, mt * 16, kt * 32, lane);
      m0 = wmma_bf16(a, ldg_b_frag(wf2_pk, 16, kt, ntb + 0, lane), m0);
      m1 = wmma_bf16(a, ldg_b_frag(wf2_pk, 16, kt, ntb + 1, lane), m1);
      m2 = wmma_bf16(a, ldg_b_frag(wf2_pk, 16, kt, ntb + 2, lane), m2);
      m3 = wmma_bf16(a, ldg_b_frag(wf2_pk, 16, kt, ntb + 3, lane), m3);
    }
#pragma unroll
    for (int t = 0; t < 4; ++t) {
      v8f acc = (t == 0) ? m0 : (t == 1) ? m1 : (t == 2) ? m2 : m3;
      int col = (ntb + t) * 16 + cl;
      float bb = bf2p[col];
#pragma unroll
      for (int i = 0; i < 8; ++i) {
        int rb = b0 + mt * 16 + i + 8 * h;
        if (rb < BB) out[(size_t)rb * EMBD + col] = acc[i] + bb;
      }
    }
  }
}

__global__ __launch_bounds__(256) void copy_batch_kernel(
    const int* __restrict__ bn, float* __restrict__ out) {
  int i = blockIdx.x * blockDim.x + threadIdx.x;
  if (i < NN) out[i] = (float)bn[i];
}

// ---------- host launcher ----------
extern "C" void kernel_launch(void* const* d_in, const int* in_sizes, int n_in,
                              void* d_out, int out_size, void* d_ws,
                              size_t ws_size, hipStream_t stream) {
  const float* h_node = (const float*)d_in[0];
  const float* pos    = (const float*)d_in[1];
  const float* h_edge = (const float*)d_in[2];
  const float* W_node = (const float*)d_in[3];
  const float* W_edge = (const float*)d_in[4];
  const float* W_eu   = (const float*)d_in[5];
  const float* b_eu   = (const float*)d_in[6];
  const float* W_msg  = (const float*)d_in[7];
  const float* b_msg  = (const float*)d_in[8];
  const float* W_nu   = (const float*)d_in[9];
  const float* b_nu   = (const float*)d_in[10];
  const float* Wf1    = (const float*)d_in[11];
  const float* bf1p   = (const float*)d_in[12];
  const float* Wf2    = (const float*)d_in[13];
  const float* bf2p   = (const float*)d_in[14];
  const int* edge_index = (const int*)d_in[15];
  const int* batch_node = (const int*)d_in[16];
  const int* batch_edge = (const int*)d_in[17];
  const int* src = edge_index;
  const int* dstv = edge_index + EE;

  char* ws = (char*)d_ws;
  size_t off = 0;
  auto carve = [&](size_t bytes) -> void* {
    void* p = ws + off;
    off = (off + bytes + 255) & ~(size_t)255;
    return p;
  };
  float*  hnf    = (float*)carve((size_t)NN * NDIM * 4);
  __bf16* hnb    = (__bf16*)carve((size_t)NN * NDIM * 2);
  float*  hef    = (float*)carve((size_t)EE * EDIM * 4);
  __bf16* heb    = (__bf16*)carve((size_t)EE * EDIM * 2);
  float*  distg  = (float*)carve((size_t)EE * 4);
  float*  agg    = (float*)carve((size_t)NN * NDIM * 4);
  float*  poolb  = (float*)carve((size_t)BB * 384 * 4);
  float*  cntn   = (float*)carve((size_t)BB * 4);
  float*  cnte   = (float*)carve((size_t)BB * 4);
  __bf16* weu_pk  = (__bf16*)carve((size_t)LYRS * 640 * 128 * 2);
  __bf16* wmsg_pk = (__bf16*)carve((size_t)LYRS * 384 * 256 * 2);
  __bf16* wnu_pk  = (__bf16*)carve((size_t)LYRS * 512 * 256 * 2);
  __bf16* wf1_pk  = (__bf16*)carve((size_t)384 * 512 * 2);
  __bf16* wf2_pk  = (__bf16*)carve((size_t)512 * 256 * 2);

  // pack weights to bf16 WMMA fragment order
  for (int l = 0; l < LYRS; ++l) {
    pack_w_kernel<<<(20 * 8 * 32 + 255) / 256, 256, 0, stream>>>(
        W_eu + (size_t)l * 641 * 128, weu_pk + (size_t)l * 640 * 128, 640, 128);
    pack_w_kernel<<<(12 * 16 * 32 + 255) / 256, 256, 0, stream>>>(
        W_msg + (size_t)l * 384 * 256, wmsg_pk + (size_t)l * 384 * 256, 384, 256);
    pack_w_kernel<<<(16 * 16 * 32 + 255) / 256, 256, 0, stream>>>(
        W_nu + (size_t)l * 512 * 256, wnu_pk + (size_t)l * 512 * 256, 512, 256);
  }
  pack_w_kernel<<<(12 * 32 * 32 + 255) / 256, 256, 0, stream>>>(Wf1, wf1_pk, 384, 512);
  pack_w_kernel<<<(16 * 16 * 32 + 255) / 256, 256, 0, stream>>>(Wf2, wf2_pk, 512, 256);

  // embeddings + geometric feature
  embed_node_kernel<<<NN / 32, 256, 0, stream>>>(h_node, W_node, hnf, hnb);
  embed_edge_kernel<<<EE / 32, 256, 0, stream>>>(h_edge, W_edge, pos, src, dstv,
                                                 hef, heb, distg);

  // message-passing blocks
  for (int l = 0; l < LYRS; ++l) {
    (void)hipMemsetAsync(agg, 0, (size_t)NN * NDIM * 4, stream);
    edge_layer_kernel<<<EE / 32, 256, 0, stream>>>(
        hnb, heb, hef, distg, src, dstv,
        weu_pk + (size_t)l * 640 * 128,
        W_eu + (size_t)l * 641 * 128 + (size_t)640 * 128,  // dist row
        b_eu + l * 128,
        wmsg_pk + (size_t)l * 384 * 256, b_msg + l * 256, agg);
    node_layer_kernel<<<NN / 32, 256, 0, stream>>>(
        hnf, hnb, agg, wnu_pk + (size_t)l * 512 * 256, b_nu + l * 256);
  }

  // pooling + final MLP
  (void)hipMemsetAsync(poolb, 0, (size_t)BB * 384 * 4, stream);
  (void)hipMemsetAsync(cntn, 0, (size_t)BB * 4, stream);
  (void)hipMemsetAsync(cnte, 0, (size_t)BB * 4, stream);
  pool_node_kernel<<<((size_t)NN * 64 + 255) / 256, 256, 0, stream>>>(hnf, batch_node,
                                                                      poolb, cntn);
  pool_edge_kernel<<<((size_t)EE * 32 + 255) / 256, 256, 0, stream>>>(hef, batch_edge,
                                                                      poolb, cnte);
  final_mlp_kernel<<<(BB + 31) / 32, 256, 0, stream>>>(poolb, cntn, cnte, wf1_pk,
                                                       bf1p, wf2_pk, bf2p,
                                                       (float*)d_out);
  if (out_size >= BB * EMBD + NN)
    copy_batch_kernel<<<(NN + 255) / 256, 256, 0, stream>>>(
        batch_node, (float*)d_out + (size_t)BB * EMBD);
}
